// Decoder_9199819948084
// MI455X (gfx1250) — compile-verified, matrix operand order checked by reading, not run on hardware
//
#include <hip/hip_runtime.h>

// ---------------------------------------------------------------------------
// Decoder step: attention (bf16 WMMA energy GEMM) + softmax + context + LSTM
// B=32, S=4096, H=O=512
// ---------------------------------------------------------------------------

#define Bn   32
#define Sn   4096
#define Hn   512

typedef __attribute__((ext_vector_type(16))) __bf16 v16bf;
typedef __attribute__((ext_vector_type(8)))  float  v8f;

union Frag {
    uint4 q[2];
    v16bf v;
};

__device__ __forceinline__ unsigned int f2bf(float f) {
    unsigned int u = __float_as_uint(f);
    u += 0x7FFFu + ((u >> 16) & 1u);   // round to nearest even
    return u >> 16;
}
__device__ __forceinline__ unsigned int pk2(float lo, float hi) {
    return f2bf(lo) | (f2bf(hi) << 16);
}
__device__ __forceinline__ float sigmoidf(float x) {
    return 1.0f / (1.0f + __expf(-x));
}

// ---------------------------------------------------------------------------
// Pack W2 (= attn_W[:, H:2H]) into CDNA5 WMMA B-matrix tiles (32x16 bf16).
// B[k][n] = W2[n][k].  Tile id = kt*32+nt, per lane 8 packed dwords:
//   n = nt*16 + lane%16 ; k = kt*32 + (lane>=16 ? 16 : 0) + 2j,2j+1
// ---------------------------------------------------------------------------
__global__ __launch_bounds__(256) void packW2_kernel(const float* __restrict__ attn_W,
                                                     unsigned int* __restrict__ w2p) {
    int t    = blockIdx.x * 256 + threadIdx.x;   // 0 .. 16383  (512 tiles * 32 lanes)
    int tile = t >> 5;
    int lane = t & 31;
    int kt = tile >> 5;
    int nt = tile & 31;
    int n  = nt * 16 + (lane & 15);
    int kb = kt * 32 + ((lane >> 4) << 4);
    const float* src = attn_W + (size_t)n * (2 * Hn) + Hn + kb;   // W2 row n
    unsigned int* dst = w2p + (size_t)t * 8;
#pragma unroll
    for (int j = 0; j < 8; ++j)
        dst[j] = pk2(src[2 * j], src[2 * j + 1]);
}

// ---------------------------------------------------------------------------
// hq[b,g] = attn_b[g] + dot(hidden[b,:], W1[g,:])      (W1 = attn_W[:, :H])
// ---------------------------------------------------------------------------
__global__ __launch_bounds__(512) void hq_kernel(const float* __restrict__ hidden,
                                                 const float* __restrict__ attn_W,
                                                 const float* __restrict__ attn_b,
                                                 float* __restrict__ hq) {
    __shared__ float hrow[Hn];
    int b = blockIdx.x, tid = threadIdx.x;
    hrow[tid] = hidden[b * Hn + tid];
    __syncthreads();
    float acc = attn_b[tid];
    const float4* wr = (const float4*)(attn_W + (size_t)tid * (2 * Hn));
#pragma unroll 4
    for (int k = 0; k < Hn / 4; ++k) {
        float4 wv = wr[k];
        const float* hp = &hrow[k * 4];
        acc += wv.x * hp[0] + wv.y * hp[1] + wv.z * hp[2] + wv.w * hp[3];
    }
    hq[b * Hn + tid] = acc;
}

// ---------------------------------------------------------------------------
// Fused energy GEMM + tanh + (.v) contraction -> raw scores[b,s].
// One block = (b, 32 rows of s) = two 16-row M-tiles staged bf16 in LDS.
// Each of 16 waves computes a 2x2 tile grid (2 M-tiles x 2 N-tiles):
//   4 accumulators, 4 wmma per K-step sharing 2 A-frags + 2 B-frags,
//   with next K-step fragments prefetched before the current WMMAs issue.
// ---------------------------------------------------------------------------
__global__ __launch_bounds__(512) void attn_energy_kernel(const float* __restrict__ enc,
                                                          const unsigned int* __restrict__ w2p,
                                                          const float* __restrict__ hq,
                                                          const float* __restrict__ vvec,
                                                          float* __restrict__ scores) {
    __shared__ unsigned int ldsA[32 * 256];   // 32 rows x 512 bf16 (packed pairs) = 32 KB
    __shared__ float sred[32];

    const int bx  = blockIdx.x;
    const int b   = bx >> 7;              // S/32 = 128 row-tiles per batch
    const int s0  = (bx & 127) << 5;
    const int tid = threadIdx.x;
    const int lane = tid & 31;
    const int wave = tid >> 5;

    // Stage + convert: 32 rows x 32 segments of 16 floats; 2 tasks per thread.
#pragma unroll
    for (int t = 0; t < 2; ++t) {
        int task = tid + t * 512;         // 0..1023
        int row  = task >> 5;             // 0..31
        int seg  = task & 31;             // 0..31
        const float4* src =
            (const float4*)(enc + ((size_t)(b * Sn + s0 + row)) * Hn + seg * 16);
        float4 f0 = src[0], f1 = src[1], f2 = src[2], f3 = src[3];
        unsigned int* dst = &ldsA[row * 256 + seg * 8];
        dst[0] = pk2(f0.x, f0.y); dst[1] = pk2(f0.z, f0.w);
        dst[2] = pk2(f1.x, f1.y); dst[3] = pk2(f1.z, f1.w);
        dst[4] = pk2(f2.x, f2.y); dst[5] = pk2(f2.z, f2.w);
        dst[6] = pk2(f3.x, f3.y); dst[7] = pk2(f3.z, f3.w);
    }
    if (tid < 32) sred[tid] = 0.0f;
    __syncthreads();

    const int col  = lane & 15;   // N col / A row index this lane carries
    const int half = lane >> 4;   // K-half for A, row-half for D

    v8f acc00 = {}, acc01 = {}, acc10 = {}, acc11 = {};
    const int nt0 = wave * 2;
    const int nt1 = wave * 2 + 1;

    const unsigned int* ldsRow0 = &ldsA[col * 256 + half * 8];          // M-tile 0
    const unsigned int* ldsRow1 = &ldsA[(16 + col) * 256 + half * 8];   // M-tile 1
    const unsigned int* bBase0  = w2p + (((size_t)nt0 * 32 + lane) << 3);
    const unsigned int* bBase1  = w2p + (((size_t)nt1 * 32 + lane) << 3);

    Frag a0, a1, bf0, bf1;
    {   // prologue: K-step 0 fragments
        const uint4* ap0 = (const uint4*)ldsRow0;
        const uint4* ap1 = (const uint4*)ldsRow1;
        a0.q[0] = ap0[0]; a0.q[1] = ap0[1];
        a1.q[0] = ap1[0]; a1.q[1] = ap1[1];
        const uint4* bp0 = (const uint4*)bBase0;
        const uint4* bp1 = (const uint4*)bBase1;
        bf0.q[0] = bp0[0]; bf0.q[1] = bp0[1];
        bf1.q[0] = bp1[0]; bf1.q[1] = bp1[1];
    }

#pragma unroll
    for (int kt = 0; kt < 16; ++kt) {
        Frag na0, na1, nb0, nb1;
        if (kt < 15) {   // prefetch next K-step fragments (resolved at unroll time)
            const uint4* ap0 = (const uint4*)(ldsRow0 + (kt + 1) * 16);
            const uint4* ap1 = (const uint4*)(ldsRow1 + (kt + 1) * 16);
            na0.q[0] = ap0[0]; na0.q[1] = ap0[1];
            na1.q[0] = ap1[0]; na1.q[1] = ap1[1];
            const uint4* bp0 = (const uint4*)(bBase0 + (size_t)(kt + 1) * 32 * 32 * 8);
            const uint4* bp1 = (const uint4*)(bBase1 + (size_t)(kt + 1) * 32 * 32 * 8);
            nb0.q[0] = bp0[0]; nb0.q[1] = bp0[1];
            nb1.q[0] = bp1[0]; nb1.q[1] = bp1[1];
        }
        acc00 = __builtin_amdgcn_wmma_f32_16x16x32_bf16(false, a0.v, false, bf0.v,
                                                        (short)0, acc00, false, false);
        acc01 = __builtin_amdgcn_wmma_f32_16x16x32_bf16(false, a0.v, false, bf1.v,
                                                        (short)0, acc01, false, false);
        acc10 = __builtin_amdgcn_wmma_f32_16x16x32_bf16(false, a1.v, false, bf0.v,
                                                        (short)0, acc10, false, false);
        acc11 = __builtin_amdgcn_wmma_f32_16x16x32_bf16(false, a1.v, false, bf1.v,
                                                        (short)0, acc11, false, false);
        if (kt < 15) {
            a0 = na0; a1 = na1; bf0 = nb0; bf1 = nb1;
        }
    }

    // Epilogue: tanh(acc + hq[g]) * v[g], reduce over g within each 16-lane half.
    auto epilogue = [&](v8f& acc, int nt, int mrow) {
        int g = nt * 16 + col;
        float hqv = hq[b * Hn + g];
        float vv  = vvec[g];
#pragma unroll
        for (int r = 0; r < 8; ++r) {
            float e = tanhf(acc[r] + hqv) * vv;
            e += __shfl_xor(e, 1, 32);
            e += __shfl_xor(e, 2, 32);
            e += __shfl_xor(e, 4, 32);
            e += __shfl_xor(e, 8, 32);
            if (col == 0) atomicAdd(&sred[mrow + r + half * 8], e);
        }
    };
    epilogue(acc00, nt0, 0);
    epilogue(acc01, nt1, 0);
    epilogue(acc10, nt0, 16);
    epilogue(acc11, nt1, 16);
    __syncthreads();

    if (tid < 32) scores[(size_t)b * Sn + s0 + tid] = sred[tid];
}

// ---------------------------------------------------------------------------
// Softmax over S per batch row.
// ---------------------------------------------------------------------------
__global__ __launch_bounds__(256) void softmax_kernel(const float* __restrict__ scores,
                                                      float* __restrict__ attn) {
    __shared__ float red[256];
    int b = blockIdx.x, tid = threadIdx.x;
    const float* srow = scores + (size_t)b * Sn;

    float m = -1e30f;
    for (int i = tid; i < Sn; i += 256) m = fmaxf(m, srow[i]);
    red[tid] = m;
    __syncthreads();
    for (int o = 128; o > 0; o >>= 1) {
        if (tid < o) red[tid] = fmaxf(red[tid], red[tid + o]);
        __syncthreads();
    }
    m = red[0];
    __syncthreads();

    float s = 0.0f;
    for (int i = tid; i < Sn; i += 256) s += __expf(srow[i] - m);
    red[tid] = s;
    __syncthreads();
    for (int o = 128; o > 0; o >>= 1) {
        if (tid < o) red[tid] += red[tid + o];
        __syncthreads();
    }
    float inv = 1.0f / red[0];

    float* arow = attn + (size_t)b * Sn;
    for (int i = tid; i < Sn; i += 256) arow[i] = __expf(srow[i] - m) * inv;
}

__global__ __launch_bounds__(256) void zero_kernel(float* __restrict__ p, int n) {
    int i = blockIdx.x * 256 + threadIdx.x;
    if (i < n) p[i] = 0.0f;
}

// ---------------------------------------------------------------------------
// context[b,h] = sum_s attn[b,s] * enc[b,s,h]   (s-chunked, global atomics)
// ---------------------------------------------------------------------------
__global__ __launch_bounds__(256) void context_kernel(const float* __restrict__ enc,
                                                      const float* __restrict__ attn,
                                                      float* __restrict__ ctx) {
    int b = blockIdx.y, chunk = blockIdx.x, tid = threadIdx.x;
    int sBase = chunk * 256;
    const float* erow = enc + ((size_t)b * Sn + sBase) * Hn;
    const float* w    = attn + (size_t)b * Sn + sBase;
    float a0 = 0.0f, a1 = 0.0f;
    for (int s = 0; s < 256; ++s) {
        float ww = w[s];
        a0 += ww * erow[(size_t)s * Hn + tid];
        a1 += ww * erow[(size_t)s * Hn + tid + 256];
    }
    atomicAdd(&ctx[b * Hn + tid], a0);
    atomicAdd(&ctx[b * Hn + tid + 256], a1);
}

// ---------------------------------------------------------------------------
// gates[b,g] = b_ih[g] + b_hh[g] + dot([ctx[b],dec[b]], W_ih[g,:])
// ---------------------------------------------------------------------------
__global__ __launch_bounds__(256) void gates_kernel(const float* __restrict__ ctx,
                                                    const float* __restrict__ dec,
                                                    const float* __restrict__ W_ih,
                                                    const float* __restrict__ b_ih,
                                                    const float* __restrict__ b_hh,
                                                    float* __restrict__ gates) {
    __shared__ float x[2 * Hn];
    int b = blockIdx.x >> 3, chunk = blockIdx.x & 7, tid = threadIdx.x;
    for (int i = tid; i < 2 * Hn; i += 256)
        x[i] = (i < Hn) ? ctx[b * Hn + i] : dec[b * Hn + (i - Hn)];
    __syncthreads();
    int g = chunk * 256 + tid;
    float acc = b_ih[g] + b_hh[g];
    const float4* wr = (const float4*)(W_ih + (size_t)g * (2 * Hn));
#pragma unroll 4
    for (int k = 0; k < (2 * Hn) / 4; ++k) {
        float4 wv = wr[k];
        const float* xp = &x[k * 4];
        acc += wv.x * xp[0] + wv.y * xp[1] + wv.z * xp[2] + wv.w * xp[3];
    }
    gates[b * (4 * Hn) + g] = acc;
}

// ---------------------------------------------------------------------------
// LSTM elementwise + output projection (h staged in LDS).
// out layout: [0:16384) output, [16384:32768) post_hidden, [32768:) attn
// ---------------------------------------------------------------------------
__global__ __launch_bounds__(512) void lstm_out_kernel(const float* __restrict__ gates,
                                                       const float* __restrict__ out_W,
                                                       const float* __restrict__ out_b,
                                                       float* __restrict__ out) {
    __shared__ float h[Hn];
    int b = blockIdx.x, tid = threadIdx.x;
    const float* gr = gates + (size_t)b * (4 * Hn);
    float gi = gr[tid];
    float gg = gr[2 * Hn + tid];
    float go = gr[3 * Hn + tid];
    float c  = sigmoidf(gi) * tanhf(gg);
    float hv = sigmoidf(go) * tanhf(c);
    h[tid] = hv;
    out[Bn * Hn + b * Hn + tid] = hv;   // post_hidden
    __syncthreads();
    float acc = out_b[tid];
    const float4* wr = (const float4*)(out_W + (size_t)tid * Hn);
#pragma unroll 4
    for (int k = 0; k < Hn / 4; ++k) {
        float4 wv = wr[k];
        const float* hp = &h[k * 4];
        acc += wv.x * hp[0] + wv.y * hp[1] + wv.z * hp[2] + wv.w * hp[3];
    }
    out[b * Hn + tid] = acc;            // output
}

// ---------------------------------------------------------------------------
extern "C" void kernel_launch(void* const* d_in, const int* in_sizes, int n_in,
                              void* d_out, int out_size, void* d_ws, size_t ws_size,
                              hipStream_t stream) {
    const float* dec     = (const float*)d_in[0];
    const float* hidden  = (const float*)d_in[1];
    const float* enc     = (const float*)d_in[2];
    const float* attn_W  = (const float*)d_in[3];
    const float* attn_b  = (const float*)d_in[4];
    const float* v       = (const float*)d_in[5];
    const float* W_ih    = (const float*)d_in[6];
    // d_in[7] = W_hh : dead in the reference (gates use W_ih only)
    const float* b_ih    = (const float*)d_in[8];
    const float* b_hh    = (const float*)d_in[9];
    const float* out_W   = (const float*)d_in[10];
    const float* out_b   = (const float*)d_in[11];
    float* out = (float*)d_out;

    char* ws = (char*)d_ws;
    unsigned int* w2p = (unsigned int*)ws;                 // 512 KB packed bf16 W2 tiles
    float* hq     = (float*)(ws + (512u << 10));           //  64 KB
    float* scores = (float*)(ws + (576u << 10));           // 512 KB
    float* ctx    = (float*)(ws + (1088u << 10));          //  64 KB
    float* gates  = (float*)(ws + (1152u << 10));          // 256 KB

    float* attn = out + 2 * Bn * Hn;                       // attn_weight output region

    packW2_kernel<<<64, 256, 0, stream>>>(attn_W, w2p);
    hq_kernel<<<Bn, Hn, 0, stream>>>(hidden, attn_W, attn_b, hq);
    attn_energy_kernel<<<Bn * (Sn / 32), 512, 0, stream>>>(enc, w2p, hq, v, scores);
    softmax_kernel<<<Bn, 256, 0, stream>>>(scores, attn);
    zero_kernel<<<(Bn * Hn + 255) / 256, 256, 0, stream>>>(ctx, Bn * Hn);
    context_kernel<<<dim3(Sn / 256, Bn), 256, 0, stream>>>(enc, attn, ctx);
    gates_kernel<<<Bn * 8, 256, 0, stream>>>(ctx, dec, W_ih, b_ih, b_hh, gates);
    lstm_out_kernel<<<Bn, Hn, 0, stream>>>(gates, out_W, out_b, out);
}
